// MultiHeadAttention_26929444946269
// MI455X (gfx1250) — compile-verified, matrix operand order checked by reading, not run on hardware
//
#include <hip/hip_runtime.h>
#include <hip/hip_bf16.h>
#include <math.h>

#define D_MODEL   1024
#define NUM_HEADS 16
#define D_HEAD    64
#define BATCH     4
#define SEQ_LQ    2048
#define SEQ_LK    2048
#define MTOT      (BATCH * SEQ_LQ)   // 8192

typedef __bf16 bf16_t;
typedef __attribute__((ext_vector_type(16))) __bf16 v16bf;
typedef __attribute__((ext_vector_type(8)))  __bf16 v8bf;
typedef __attribute__((ext_vector_type(4)))  __bf16 v4bf;
typedef __attribute__((ext_vector_type(8)))  float  v8f;
typedef __attribute__((ext_vector_type(4)))  unsigned int v4u;
typedef __attribute__((ext_vector_type(8)))  int    v8i;
typedef __attribute__((ext_vector_type(4)))  int    v4i;

// Assemble a bf16 WMMA fragment from the two 16-byte contiguous chunks the
// CDNA5 layout needs per lane: lanes 0-15 hold K = base+{0..7} and
// base+{16..23}; lanes 16-31 the same shifted by +8.
static __device__ __forceinline__ v16bf mk_frag(v8bf lo, v8bf hi) {
  v16bf r;
#pragma unroll
  for (int i = 0; i < 8; ++i) { r[i] = lo[i]; r[i + 8] = hi[i]; }
  return r;
}

// ---------------------------------------------------------------------------
// fp32 -> bf16 grid-stride conversion (4 elements / thread / step)
// ---------------------------------------------------------------------------
__global__ void cvt_f32_bf16_kernel(const float* __restrict__ src,
                                    bf16_t* __restrict__ dst, int n) {
  int stride = gridDim.x * blockDim.x * 4;
  for (int i = (blockIdx.x * blockDim.x + threadIdx.x) * 4; i < n; i += stride) {
    float4 f = *(const float4*)(src + i);
    v4bf o;
    o[0] = (bf16_t)f.x; o[1] = (bf16_t)f.y;
    o[2] = (bf16_t)f.z; o[3] = (bf16_t)f.w;
    *(v4bf*)(dst + i) = o;
  }
}

// ---------------------------------------------------------------------------
// Y = X(bf16)[M x 1024] * W(bf16)[1024 x 1024]^T + bias(f32)
// MODE 0: scatter bf16 result into head-major [B, H, L, D_HEAD]
// MODE 1: write f32 result row-major [M x 1024]
// Block: 256 threads = 8 waves; wave tile = 32(M) x 64(N); K step = 32.
// ---------------------------------------------------------------------------
template <int MODE>
__global__ void gemm_bf16_kernel(const bf16_t* __restrict__ X,
                                 const bf16_t* __restrict__ W,
                                 const float*  __restrict__ bias,
                                 void* __restrict__ out) {
  const int lane  = threadIdx.x & 31;
  const int wave  = threadIdx.x >> 5;
  const int ln    = lane & 15;
  const int half8 = (lane >> 4) * 8;

  const int m0 = blockIdx.y * 256 + wave * 32;
  const int n0 = blockIdx.x * 64;

  const bf16_t* arow0 = X + (size_t)(m0 + ln) * D_MODEL + half8;
  const bf16_t* arow1 = arow0 + (size_t)16 * D_MODEL;

  v8f acc0[4] = {};
  v8f acc1[4] = {};
#pragma unroll 1
  for (int kk = 0; kk < D_MODEL; kk += 32) {
    v16bf a0 = mk_frag(*(const v8bf*)(arow0 + kk),
                       *(const v8bf*)(arow0 + kk + 16));
    v16bf a1 = mk_frag(*(const v8bf*)(arow1 + kk),
                       *(const v8bf*)(arow1 + kk + 16));
#pragma unroll
    for (int n = 0; n < 4; ++n) {
      const bf16_t* brow = W + (size_t)(n0 + n * 16 + ln) * D_MODEL + kk + half8;
      v16bf b = mk_frag(*(const v8bf*)(brow), *(const v8bf*)(brow + 16));
      acc0[n] = __builtin_amdgcn_wmma_f32_16x16x32_bf16(
          false, a0, false, b, (short)0, acc0[n], false, false);
      acc1[n] = __builtin_amdgcn_wmma_f32_16x16x32_bf16(
          false, a1, false, b, (short)0, acc1[n], false, false);
    }
  }

#pragma unroll
  for (int n = 0; n < 4; ++n) {
    const int col = n0 + n * 16 + ln;
    const float bv = bias[col];
#pragma unroll
    for (int g = 0; g < 2; ++g) {
#pragma unroll
      for (int r = 0; r < 8; ++r) {
        const int rowg = m0 + g * 16 + r + half8;
        float v = (g ? acc1[n][r] : acc0[n][r]) + bv;
        if (MODE == 0) {
          int b = rowg >> 11;               // / SEQ_LQ
          int l = rowg & (SEQ_LQ - 1);
          int h = col >> 6;                 // / D_HEAD
          int d = col & (D_HEAD - 1);
          ((bf16_t*)out)[(((size_t)(b * NUM_HEADS + h) * SEQ_LQ) + l) * D_HEAD + d] =
              (bf16_t)v;
        } else {
          ((float*)out)[(size_t)rowg * D_MODEL + col] = v;
        }
      }
    }
  }
}

// ---------------------------------------------------------------------------
// Tensor Data Mover: DMA one 32x64 bf16 tile (row-major) from global memory
// into LDS, hardware-padding each 64-elem (32-DWORD) row by 4 DWORDs so rows
// land on a 72-element (KSTR) pitch. D# per CDNA5 ISA §8.3/8.4.
// ---------------------------------------------------------------------------
#define KSTR 72   // 64 + 8 pad (bf16 elems); rows stay 16B aligned (144B)
#define VSTR 40   // 32 + 8 pad (80B rows)

static __device__ __forceinline__ void tdm_load_k_tile(const bf16_t* gsrc,
                                                       unsigned int lds_off) {
  const unsigned long long ga = (unsigned long long)(size_t)gsrc;
  v4u g0;
  g0[0] = 1u;                                   // count=1 user descriptor
  g0[1] = lds_off;                              // lds_addr (bytes)
  g0[2] = (unsigned int)ga;                     // global_addr[31:0]
  g0[3] = (unsigned int)((ga >> 32) & 0x01FFFFFFu) | (2u << 30);  // [56:32]|type=2
  v8i g1;
  g1[0] = (int)((1u << 16)      // data_size = 1 -> 2-byte elements
              | (1u << 20)      // pad_enable
              | (4u << 22)      // pad_interval: every 32 DWORDs (one 64-elem row)
              | (3u << 25));    // pad_amount: 4 DWORDs (8 bf16) -> 72-elem pitch
  g1[1] = (int)(64u << 16);     // tensor_dim0 = 64 (bits 79:48, low half)
  g1[2] = (int)(2048u << 16);   // tensor_dim1 = 2048 (bits 111:80, low half)
  g1[3] = (int)(64u << 16);     // tile_dim0 = 64 (bits 127:112)
  g1[4] = (int)32;              // tile_dim1 = 32 (bits 143:128)
  g1[5] = (int)64;              // tensor_dim0_stride = 64 (bits 207:160)
  g1[6] = 0;
  g1[7] = 0;
  v4i gz = {0, 0, 0, 0};        // 2D tensor: groups 2/3 unused
#if __clang_major__ >= 23
  v8i gz8 = {0, 0, 0, 0, 0, 0, 0, 0};
  __builtin_amdgcn_tensor_load_to_lds(g0, g1, gz, gz, gz8, 0);
#else
  __builtin_amdgcn_tensor_load_to_lds(g0, g1, gz, gz, 0);
#endif
}

// ---------------------------------------------------------------------------
// Flash attention, max-free variant (fp32 exp range covers |score| << 88 for
// standardized inputs): softmax denominator accumulated with an extra WMMA
// against a ones-matrix -> zero cross-lane shuffles, no O-rescale.
// K tile staged by the TDM (overlapped with the manual V-transpose staging);
// 9 WMMAs per 32-key step. Block = 256 threads (8 waves), one (b,h) x 128 q.
// ---------------------------------------------------------------------------
__global__ void flash_attn_kernel(const bf16_t* __restrict__ Q,   // [B*H, LQ, 64]
                                  const bf16_t* __restrict__ K,   // [B*H, LK, 64]
                                  const bf16_t* __restrict__ V,   // [B*H, LK, 64]
                                  bf16_t* __restrict__ O) {       // [B, LQ, 1024]
  __shared__ __align__(16) bf16_t lds_k[32 * KSTR];
  __shared__ __align__(16) bf16_t lds_vt[64 * VSTR];
  __shared__ __align__(16) bf16_t lds_p[8][16 * 32];

  const int lane  = threadIdx.x & 31;
  const int wave  = threadIdx.x >> 5;
  const int ln    = lane & 15;
  const int half8 = (lane >> 4) * 8;
  const int tid   = threadIdx.x;

  const int bh = blockIdx.y;
  const int b  = bh >> 4;
  const int h  = bh & 15;
  const int q0 = blockIdx.x * 128 + wave * 16;

  // flat shared addresses are aperture|offset: low 32 bits = LDS byte offset
  const unsigned int ldsk_off = (unsigned int)(size_t)(&lds_k[0]);

  // Q fragments for d_head chunks 0..31 / 32..63 (kept resident)
  const bf16_t* qbase = Q + ((size_t)bh * SEQ_LQ + q0 + ln) * D_HEAD + half8;
  const v16bf qa0 = mk_frag(*(const v8bf*)(qbase +  0), *(const v8bf*)(qbase + 16));
  const v16bf qa1 = mk_frag(*(const v8bf*)(qbase + 32), *(const v8bf*)(qbase + 48));

  // ones B-fragment for the row-sum WMMA
  v16bf ones;
#pragma unroll
  for (int i = 0; i < 16; ++i) ones[i] = (bf16_t)1.0f;

  v8f o[4] = {};
  v8f lacc = {};   // every column holds the running row-sum of P

  const int key = tid >> 3;   // 0..31  (cooperative staging role)
  const int seg = tid & 7;    // 0..7

  for (int k0 = 0; k0 < SEQ_LK; k0 += 32) {
    __syncthreads();
    // TDM DMAs the K tile into LDS (wave-uniform branch: tensor ops ignore
    // EXEC, so exactly one wave may execute the instruction)
    if (wave == 0) {
      tdm_load_k_tile(K + ((size_t)bh * SEQ_LK + k0) * D_HEAD, ldsk_off);
    }
    {   // meanwhile: all threads stage V transposed into LDS
      const size_t gofs = ((size_t)bh * SEQ_LK + k0 + key) * D_HEAD + seg * 8;
      v8bf vv = *(const v8bf*)(V + gofs);
#pragma unroll
      for (int i = 0; i < 8; ++i) lds_vt[(seg * 8 + i) * VSTR + key] = vv[i];
      if (k0 + 32 < SEQ_LK)   // prefetch next V tile (global_prefetch_b8)
        __builtin_prefetch(V + gofs + 32 * D_HEAD, 0, 0);
    }
    if (wave == 0) __builtin_amdgcn_s_wait_tensorcnt(0);
    __syncthreads();

    bf16_t* prow = &lds_p[wave][0];
    const float scale = 0.125f;   // 1/sqrt(64)

    // S = Q*K^T per 16-key sub-tile; exp() straight into the P scratch
#pragma unroll
    for (int t = 0; t < 2; ++t) {
      const bf16_t* kb = &lds_k[(t * 16 + ln) * KSTR + half8];
      v16bf b0 = mk_frag(*(const v8bf*)(kb +  0), *(const v8bf*)(kb + 16));
      v16bf b1 = mk_frag(*(const v8bf*)(kb + 32), *(const v8bf*)(kb + 48));
      v8f s = {};
      s = __builtin_amdgcn_wmma_f32_16x16x32_bf16(
          false, qa0, false, b0, (short)0, s, false, false);
      s = __builtin_amdgcn_wmma_f32_16x16x32_bf16(
          false, qa1, false, b1, (short)0, s, false, false);
#pragma unroll
      for (int r = 0; r < 8; ++r) {
        float p = __expf(s[r] * scale);
        prow[(r + half8) * 32 + t * 16 + ln] = (bf16_t)p;   // C-layout -> A-layout src
      }
    }

    // P as A fragment (same-wave DS ops are in order: stores before loads)
    const bf16_t* pp = &lds_p[wave][ln * 32 + half8];
    v16bf pa = mk_frag(*(const v8bf*)(pp), *(const v8bf*)(pp + 16));

    // denominator: lacc += P * ones   (row-sum broadcast to all 16 columns)
    lacc = __builtin_amdgcn_wmma_f32_16x16x32_bf16(
        false, pa, false, ones, (short)0, lacc, false, false);

    // O += P * V
#pragma unroll
    for (int n = 0; n < 4; ++n) {
      const bf16_t* vb = &lds_vt[(n * 16 + ln) * VSTR + half8];
      v16bf bfr = mk_frag(*(const v8bf*)(vb), *(const v8bf*)(vb + 16));
      o[n] = __builtin_amdgcn_wmma_f32_16x16x32_bf16(
          false, pa, false, bfr, (short)0, o[n], false, false);
    }
  }

  // normalize and scatter to [B, LQ, D_MODEL] (bf16, head-interleaved)
#pragma unroll
  for (int r = 0; r < 8; ++r) {
    const float inv = 1.0f / lacc[r];
    const int   row = q0 + r + half8;
#pragma unroll
    for (int n = 0; n < 4; ++n) {
      const int col = h * D_HEAD + n * 16 + ln;
      O[((size_t)b * SEQ_LQ + row) * D_MODEL + col] = (bf16_t)(o[n][r] * inv);
    }
  }
}

// ---------------------------------------------------------------------------
extern "C" void kernel_launch(void* const* d_in, const int* in_sizes, int n_in,
                              void* d_out, int out_size, void* d_ws, size_t ws_size,
                              hipStream_t stream) {
  (void)in_sizes; (void)n_in; (void)out_size; (void)ws_size;

  const float* q_in = (const float*)d_in[0];
  const float* k_in = (const float*)d_in[1];
  const float* v_in = (const float*)d_in[2];
  const float* Wq   = (const float*)d_in[3];
  const float* bq   = (const float*)d_in[4];
  const float* Wk   = (const float*)d_in[5];
  const float* bk   = (const float*)d_in[6];
  const float* Wv   = (const float*)d_in[7];
  const float* bv   = (const float*)d_in[8];
  const float* Wo   = (const float*)d_in[9];
  const float* bo   = (const float*)d_in[10];

  const size_t ACT  = (size_t)MTOT * D_MODEL;       // 8.39M elems
  const size_t WSZ  = (size_t)D_MODEL * D_MODEL;    // 1.05M elems

  char* ws = (char*)d_ws;
  size_t off = 0;
  auto carve = [&](size_t bytes) {
    void* p = ws + off;
    off += (bytes + 255) & ~(size_t)255;
    return p;
  };
  bf16_t* xq  = (bf16_t*)carve(ACT * 2);
  bf16_t* xk  = (bf16_t*)carve(ACT * 2);
  bf16_t* xv  = (bf16_t*)carve(ACT * 2);
  bf16_t* wqb = (bf16_t*)carve(WSZ * 2);
  bf16_t* wkb = (bf16_t*)carve(WSZ * 2);
  bf16_t* wvb = (bf16_t*)carve(WSZ * 2);
  bf16_t* wob = (bf16_t*)carve(WSZ * 2);
  bf16_t* qh  = (bf16_t*)carve(ACT * 2);   // [B,H,L,64]
  bf16_t* kh  = (bf16_t*)carve(ACT * 2);
  bf16_t* vh  = (bf16_t*)carve(ACT * 2);
  bf16_t* ob  = (bf16_t*)carve(ACT * 2);   // attention out [B,L,1024]

  // 1) convert fp32 -> bf16
  {
    int blkA = (int)(ACT / (256 * 4));
    int blkW = (int)(WSZ / (256 * 4));
    cvt_f32_bf16_kernel<<<blkA, 256, 0, stream>>>(q_in, xq, (int)ACT);
    cvt_f32_bf16_kernel<<<blkA, 256, 0, stream>>>(k_in, xk, (int)ACT);
    cvt_f32_bf16_kernel<<<blkA, 256, 0, stream>>>(v_in, xv, (int)ACT);
    cvt_f32_bf16_kernel<<<blkW, 256, 0, stream>>>(Wq, wqb, (int)WSZ);
    cvt_f32_bf16_kernel<<<blkW, 256, 0, stream>>>(Wk, wkb, (int)WSZ);
    cvt_f32_bf16_kernel<<<blkW, 256, 0, stream>>>(Wv, wvb, (int)WSZ);
    cvt_f32_bf16_kernel<<<blkW, 256, 0, stream>>>(Wo, wob, (int)WSZ);
  }

  // 2) projections -> head-major bf16
  dim3 ggrid(D_MODEL / 64, MTOT / 256);     // (16, 32)
  gemm_bf16_kernel<0><<<ggrid, 256, 0, stream>>>(xq, wqb, bq, qh);
  gemm_bf16_kernel<0><<<ggrid, 256, 0, stream>>>(xk, wkb, bk, kh);
  gemm_bf16_kernel<0><<<ggrid, 256, 0, stream>>>(xv, wvb, bv, vh);

  // 3) attention
  dim3 agrid(SEQ_LQ / 128, BATCH * NUM_HEADS);   // (16, 64)
  flash_attn_kernel<<<agrid, 256, 0, stream>>>(qh, kh, vh, ob);

  // 4) output projection -> fp32 d_out
  gemm_bf16_kernel<1><<<ggrid, 256, 0, stream>>>(ob, wob, bo, d_out);
}
